// Net_25778393710796
// MI455X (gfx1250) — matrix operand order repaired and checked by reading, not compiled, not run on hardware
//
#include <hip/hip_runtime.h>

// GCN 2-layer: h = relu(GCNConv(x, W1, b1)); out = GCNConv(h, W2, b2)
// N=50000 (multiple of 16), E=800000, D: 128 -> 256 -> 128.

typedef __attribute__((ext_vector_type(16))) __bf16 v16bf;
typedef __attribute__((ext_vector_type(8)))  float  v8f;

union BV { v16bf v; uint4 q[2]; };

static __device__ __forceinline__ unsigned short f2bf(float f) {
  __bf16 h = (__bf16)f;
  return __builtin_bit_cast(unsigned short, h);
}

// ---------------- elementwise conversions ----------------

__global__ __launch_bounds__(256)
void k_cvt_x(const float* __restrict__ x, unsigned short* __restrict__ xb, int n4) {
  int t = blockIdx.x * blockDim.x + threadIdx.x;
  if (t >= n4) return;
  float4 v = ((const float4*)x)[t];
  ushort4 o;
  o.x = f2bf(v.x); o.y = f2bf(v.y); o.z = f2bf(v.z); o.w = f2bf(v.w);
  ((ushort4*)xb)[t] = o;
}

// Wt[o*DIN + k] = bf16(W[k*DOUT + o])  (transpose so B columns are contiguous)
template<int DIN, int DOUT>
__global__ __launch_bounds__(256)
void k_cvt_wT(const float* __restrict__ W, unsigned short* __restrict__ Wt) {
  int t = blockIdx.x * blockDim.x + threadIdx.x;
  if (t >= DIN * DOUT) return;
  int o = t / DIN, k = t % DIN;
  Wt[t] = f2bf(W[k * DOUT + o]);
}

// ---------------- degree / normalization ----------------

__global__ __launch_bounds__(256)
void k_deg_init(float* __restrict__ deg, int N) {
  int i = blockIdx.x * blockDim.x + threadIdx.x;
  if (i < N) deg[i] = 1.0f;   // self-loop
}

__global__ __launch_bounds__(256)
void k_deg_count(const long long* __restrict__ ei, float* __restrict__ deg, int E) {
  int e = blockIdx.x * blockDim.x + threadIdx.x;
  if (e >= E) return;
  long long col = ei[(long long)E + e];
  atomicAdd(&deg[col], 1.0f);
}

__global__ __launch_bounds__(256)
void k_deg_rsqrt(float* __restrict__ deg, int N) {
  int i = blockIdx.x * blockDim.x + threadIdx.x;
  if (i < N) deg[i] = rsqrtf(deg[i]);  // deg >= 1 always (self loop)
}

// ---------------- bf16 WMMA GEMM: C = Xb @ Wt^T ----------------
// One wave computes a 16x16 tile; 4 waves/block cover 64 output cols.
// Epilogue writes raw product (gather source) and dinv^2-scaled init (+bias).
template<int DIN, int DOUT, bool BIAS>
__global__ __launch_bounds__(128)
void k_gemm(const unsigned short* __restrict__ Xb,
            const unsigned short* __restrict__ Wt,
            float* __restrict__ Craw, float* __restrict__ Cinit,
            const float* __restrict__ dinv, const float* __restrict__ bias) {
  const int lane  = threadIdx.x & 31;
  const int wave  = threadIdx.x >> 5;
  const int mtile = blockIdx.x;
  const int ntile = blockIdx.y * 4 + wave;
  const int mn    = lane & 15;
  const int khalf = lane >> 4;

  const __bf16* A = (const __bf16*)Xb + (size_t)(mtile * 16 + mn) * DIN;
  const __bf16* B = (const __bf16*)Wt + (size_t)(ntile * 16 + mn) * DIN;

  v8f c = {};
#pragma unroll
  for (int k0 = 0; k0 < DIN; k0 += 32) {
    BV a, b;
    // A 16x32 bf16: lane m, VGPR0-3 = K[khalf*8 .. +7], VGPR4-7 = K[16+khalf*8 .. +7]
    a.q[0] = *(const uint4*)(A + k0 + khalf * 8);
    a.q[1] = *(const uint4*)(A + k0 + 16 + khalf * 8);
    // B 32x16 bf16: lane n, VGPR0-3 = K[khalf*8 .. +7], VGPR4-7 = K[16+khalf*8 .. +7]
    b.q[0] = *(const uint4*)(B + k0 + khalf * 8);
    b.q[1] = *(const uint4*)(B + k0 + 16 + khalf * 8);
    c = __builtin_amdgcn_wmma_f32_16x16x32_bf16(false, a.v, false, b.v,
                                                (short)0, c, false, false);
  }

  const int col = ntile * 16 + mn;
  const float bv = BIAS ? bias[col] : 0.0f;
#pragma unroll
  for (int r = 0; r < 8; ++r) {
    int row = mtile * 16 + r + khalf * 8;
    float v = c[r];
    Craw[(size_t)row * DOUT + col] = v;
    float d = dinv[row];
    float iv = v * d * d;                         // self-loop contribution
    Cinit[(size_t)row * DOUT + col] = BIAS ? (iv + bv) : iv;
  }
}

// ---------------- edge gather + scatter-add (float atomics to L2) ----------------
template<int D>
__global__ __launch_bounds__(256)
void k_edges(const long long* __restrict__ ei, const float* __restrict__ S,
             const float* __restrict__ dinv, float* __restrict__ O, int E) {
  const int C4 = D / 4;
  long long tid = (long long)blockIdx.x * blockDim.x + threadIdx.x;
  long long e = tid / C4;
  if (e >= E) return;
  int c = (int)(tid % C4);
  long long r  = ei[e];
  long long cl = ei[(long long)E + e];
  float nrm = dinv[r] * dinv[cl];
  float4 v = *(const float4*)(S + (size_t)r * D + c * 4);
  float* o = O + (size_t)cl * D + c * 4;
  atomicAdd(o + 0, v.x * nrm);
  atomicAdd(o + 1, v.y * nrm);
  atomicAdd(o + 2, v.z * nrm);
  atomicAdd(o + 3, v.w * nrm);
}

// ---------------- bias + relu + f32->bf16 (feeds GEMM2) ----------------
__global__ __launch_bounds__(256)
void k_relu_cvt(const float* __restrict__ agg, const float* __restrict__ b1,
                unsigned short* __restrict__ hb, int n4 /* = N*256/4 */) {
  int t = blockIdx.x * blockDim.x + threadIdx.x;
  if (t >= n4) return;
  int col4 = (t & 63) * 4;                 // D_HID=256 -> 64 float4 per row
  float4 v  = ((const float4*)agg)[t];
  float4 bb = *(const float4*)(b1 + col4);
  ushort4 o;
  o.x = f2bf(fmaxf(v.x + bb.x, 0.0f));
  o.y = f2bf(fmaxf(v.y + bb.y, 0.0f));
  o.z = f2bf(fmaxf(v.z + bb.z, 0.0f));
  o.w = f2bf(fmaxf(v.w + bb.w, 0.0f));
  ((ushort4*)hb)[t] = o;
}

// ---------------- host launcher ----------------

extern "C" void kernel_launch(void* const* d_in, const int* in_sizes, int n_in,
                              void* d_out, int out_size, void* d_ws, size_t ws_size,
                              hipStream_t stream) {
  const float*     x  = (const float*)d_in[0];
  const long long* ei = (const long long*)d_in[1];
  const float*     W1 = (const float*)d_in[2];
  const float*     b1 = (const float*)d_in[3];
  const float*     W2 = (const float*)d_in[4];
  const float*     b2 = (const float*)d_in[5];
  float*           out = (float*)d_out;

  constexpr int DIN = 128, DHID = 256, DOUT = 128;
  const int N = in_sizes[0] / DIN;   // 50000 (multiple of 16)
  const int E = in_sizes[1] / 2;     // 800000

  auto al = [](size_t s) { return (s + 255) & ~(size_t)255; };
  char* ws = (char*)d_ws;
  size_t off = 0;
  float* dinv = (float*)(ws + off);           off += al((size_t)N * 4);
  size_t xw1_off = off;
  float* xw1  = (float*)(ws + off);           off += al((size_t)N * DHID * 4);
  float* agg1 = (float*)(ws + off);           off += al((size_t)N * DHID * 4);
  unsigned short* xb  = (unsigned short*)(ws + off); off += al((size_t)N * DIN * 2);
  unsigned short* W1t = (unsigned short*)(ws + off); off += al((size_t)DIN * DHID * 2);
  unsigned short* W2t = (unsigned short*)(ws + off); off += al((size_t)DHID * DOUT * 2);
  // xw1 is dead after the layer-1 edge kernel -> reuse its 51.2MB for hb + hw2
  unsigned short* hb  = (unsigned short*)(ws + xw1_off);                       // N*256 bf16
  float*          hw2 = (float*)(ws + xw1_off + (size_t)N * DHID * 2);         // N*128 f32

  // precision prep
  {
    int n4 = N * DIN / 4;
    k_cvt_x<<<(n4 + 255) / 256, 256, 0, stream>>>(x, xb, n4);
  }
  k_cvt_wT<DIN,  DHID><<<(DIN * DHID + 255) / 256, 256, 0, stream>>>(W1, W1t);
  k_cvt_wT<DHID, DOUT><<<(DHID * DOUT + 255) / 256, 256, 0, stream>>>(W2, W2t);

  // symmetric normalization
  k_deg_init <<<(N + 255) / 256, 256, 0, stream>>>(dinv, N);
  k_deg_count<<<(E + 255) / 256, 256, 0, stream>>>(ei, dinv, E);
  k_deg_rsqrt<<<(N + 255) / 256, 256, 0, stream>>>(dinv, N);

  // layer 1: xw1 = x@W1 (raw) ; agg1 = xw1 * dinv^2 (self-loop init)
  k_gemm<DIN, DHID, false><<<dim3(N / 16, DHID / 64), 128, 0, stream>>>(
      xb, W1t, xw1, agg1, dinv, nullptr);
  {
    long long tot = (long long)E * (DHID / 4);
    unsigned int blocks = (unsigned int)((tot + 255) / 256);
    k_edges<DHID><<<blocks, 256, 0, stream>>>(ei, xw1, dinv, agg1, E);
  }
  {
    int n4 = N * DHID / 4;
    k_relu_cvt<<<(n4 + 255) / 256, 256, 0, stream>>>(agg1, b1, hb, n4);
  }

  // layer 2: hw2 = h@W2 (raw) ; out = hw2 * dinv^2 + b2 (self-loop init + bias)
  k_gemm<DHID, DOUT, true><<<dim3(N / 16, DOUT / 64), 128, 0, stream>>>(
      hb, W2t, hw2, out, dinv, b2);
  {
    long long tot = (long long)E * (DOUT / 4);
    unsigned int blocks = (unsigned int)((tot + 255) / 256);
    k_edges<DOUT><<<blocks, 256, 0, stream>>>(ei, hw2, dinv, out, E);
  }
}
